// LSTMModel_74809740361860
// MI455X (gfx1250) — matrix operand order repaired and checked by reading, not compiled, not run on hardware
//
#include <hip/hip_runtime.h>
#include <hip/hip_bf16.h>

typedef __attribute__((ext_vector_type(16))) _Float16 v16h;
typedef __attribute__((ext_vector_type(8)))  _Float16 v8h;
typedef __attribute__((ext_vector_type(8)))  float    v8f;
typedef __attribute__((ext_vector_type(4)))  int      v4i;

#define B_   256
#define T_   512
#define H_   128
#define G4_  512   // 4*H
#define HSTR 136   // LDS h-tile row stride (halfs): 272B, 16B-aligned, 4-bank skew
#define GSTR 520   // LDS gate-tile row stride (halfs)
#define XSTR 520   // LDS xg-tile row stride (halfs)

#if __has_builtin(__builtin_amdgcn_global_load_async_to_lds_b128) && \
    __has_builtin(__builtin_amdgcn_s_wait_asynccnt)
#define USE_ASYNC_XG 1
#else
#define USE_ASYNC_XG 0
#endif

#if USE_ASYNC_XG
// Async global->LDS b128 copy; signature per hipcc diagnostic:
// (v4i addrspace(1)*, v4i addrspace(3)*, imm offset, imm cpol)
__device__ inline void async_copy_b128(const void* gsrc, void* ldst) {
  __builtin_amdgcn_global_load_async_to_lds_b128(
      (__attribute__((address_space(1))) v4i*)gsrc,
      (__attribute__((address_space(3))) v4i*)ldst, 0, 0);
}
#endif

// ---------------------------------------------------------------------------
// WMMA fragment loaders (ISA 7.12.2 layouts, wave32)
// ---------------------------------------------------------------------------

// A-matrix 16x32 f16 fragment. src points at a [16 x ld] f16 tile (row-major).
// lane<16:  M=lane,    K = kBase+{0..7, 16..23}
// lane>=16: M=lane-16, K = kBase+{8..15, 24..31}
__device__ inline v16h load_a_frag(const _Float16* __restrict__ src, int ld,
                                   int kBase, int lane) {
  const int m  = lane & 15;
  const int k0 = kBase + ((lane >> 4) ? 8 : 0);
  const _Float16* p = src + (size_t)m * ld + k0;
  v8h lo = *(const v8h*)(p);
  v8h hi = *(const v8h*)(p + 16);
  return __builtin_shufflevector(lo, hi, 0,1,2,3,4,5,6,7,8,9,10,11,12,13,14,15);
}

// B-matrix 32x16 f16 fragment where B(k,n) = W[n][k]  (i.e. B = W^T),
// W row-major [N x ld]. lane<16: N=lane, K=kBase+{0..7, 16..23};
// lane>=16: N=lane-16, K=kBase+{8..15, 24..31}.
__device__ inline v16h load_b_frag(const _Float16* __restrict__ W, int ld,
                                   int nBase, int kBase, int lane) {
  const int n  = lane & 15;
  const int k0 = kBase + ((lane >> 4) ? 8 : 0);
  const _Float16* p = W + (size_t)(nBase + n) * ld + k0;
  v8h lo = *(const v8h*)(p);
  v8h hi = *(const v8h*)(p + 16);
  return __builtin_shufflevector(lo, hi, 0,1,2,3,4,5,6,7,8,9,10,11,12,13,14,15);
}

__device__ inline float fast_sigmoid(float x) { return 1.0f / (1.0f + __expf(-x)); }
__device__ inline float fast_tanh(float x) {
  float e = __expf(2.0f * x);
  return 1.0f - 2.0f / (e + 1.0f);
}

// ---------------------------------------------------------------------------
// Small prep kernels
// ---------------------------------------------------------------------------
__global__ void cvt_f32_to_f16(const float* __restrict__ src,
                               _Float16* __restrict__ dst, int n) {
  int i = blockIdx.x * blockDim.x + threadIdx.x;
  if (i < n) dst[i] = (_Float16)src[i];
}

__global__ void bias_sum_kernel(const float* __restrict__ a,
                                const float* __restrict__ b,
                                float* __restrict__ out, int n) {
  int i = blockIdx.x * blockDim.x + threadIdx.x;
  if (i < n) out[i] = a[i] + b[i];
}

// ---------------------------------------------------------------------------
// Layer-0 xg: xg[t*B+b, g] = sum_{i<8} x[b,t,i]*Wih0[g,i] + bias[g]   (K=8)
// x layout is [B, T, 8]. 4 outputs per thread.
// ---------------------------------------------------------------------------
__global__ __launch_bounds__(256) void xg_l0_kernel(
    const float* __restrict__ x, const float* __restrict__ Wih,
    const float* __restrict__ bias, _Float16* __restrict__ xg) {
  size_t idx = (size_t)blockIdx.x * blockDim.x + threadIdx.x;   // over T*B*128
  if (idx >= (size_t)T_ * B_ * 128) return;
  int r  = (int)(idx >> 7);          // row = t*B + b
  int g0 = ((int)idx & 127) * 4;
  int t = r / B_, b = r % B_;
  const float* xp = x + ((size_t)b * T_ + t) * 8;
  float xv[8];
#pragma unroll
  for (int i = 0; i < 8; ++i) xv[i] = xp[i];
#pragma unroll
  for (int q = 0; q < 4; ++q) {
    int g = g0 + q;
    const float* w = Wih + g * 8;
    float acc = bias[g];
#pragma unroll
    for (int i = 0; i < 8; ++i) acc = fmaf(xv[i], w[i], acc);
    xg[(size_t)r * G4_ + g] = (_Float16)acc;
  }
}

// ---------------------------------------------------------------------------
// xg GEMM (layers 1,2): xg[R,512] = hin[R,128] @ Wih^T + bias, all f16 I/O.
// Block = 64 rows x 512 cols. 8 waves: wave&3 -> M-tile, wave>>2 -> col half.
// ---------------------------------------------------------------------------
__global__ __launch_bounds__(256) void xg_gemm_kernel(
    const _Float16* __restrict__ hin, const _Float16* __restrict__ Wih16,
    const float* __restrict__ bias, _Float16* __restrict__ xg) {
  const int lane = threadIdx.x & 31;
  const int wave = threadIdx.x >> 5;
  const int rowBase = blockIdx.x * 64 + (wave & 3) * 16;
  const int colBase = (wave >> 2) * 256;

  v16h a[4];
#pragma unroll
  for (int kc = 0; kc < 4; ++kc)
    a[kc] = load_a_frag(hin + (size_t)rowBase * H_, H_, kc * 32, lane);

  const int nlane = lane & 15;
  const int mHalf = (lane >> 4) ? 8 : 0;
#pragma unroll 4
  for (int nt = 0; nt < 16; ++nt) {
    const int nB = colBase + nt * 16;
    v8f c = {};
#pragma unroll
    for (int kc = 0; kc < 4; ++kc) {
      v16h b = load_b_frag(Wih16, H_, nB, kc * 32, lane);
      c = __builtin_amdgcn_wmma_f32_16x16x32_f16(false, a[kc], false, b,
                                                 (short)0, c, false, false);
    }
    const float bv = bias[nB + nlane];
    _Float16* outp = xg + (size_t)(rowBase + mHalf) * G4_ + nB + nlane;
#pragma unroll
    for (int r = 0; r < 8; ++r)
      outp[(size_t)r * G4_] = (_Float16)(c[r] + bv);
  }
}

// ---------------------------------------------------------------------------
// Recurrent scan: per block 16 batch rows, full time loop.
// gates[16,512] = h[16,128] @ Whh^T + xg[t]; then LSTM cell elementwise.
// Whh fragments live in VGPRs for the whole loop; h tile lives in LDS.
// xg[t+1] is prefetched into LDS with async global->LDS loads (ASYNCcnt).
// ---------------------------------------------------------------------------
__global__ __launch_bounds__(256) void lstm_scan_kernel(
    const _Float16* __restrict__ xg,     // [T, B, 512]
    const _Float16* __restrict__ Whh16,  // [512, 128]
    _Float16* __restrict__ hbuf) {       // [T, B, 128]
  __shared__ _Float16 sh_h[16 * HSTR];
  __shared__ _Float16 sh_g[16 * GSTR];
#if USE_ASYNC_XG
  __shared__ _Float16 sh_xg[2][16 * XSTR];
#endif

  const int wgRow = blockIdx.x * 16;
  const int tid  = threadIdx.x;
  const int lane = tid & 31;
  const int wave = tid >> 5;          // 0..7 -> cols [wave*64, wave*64+64)

  // Pin Whh^T fragments in registers: 4 N-tiles x 4 K-chunks per wave.
  v16h bfrag[4][4];
#pragma unroll
  for (int nt = 0; nt < 4; ++nt)
#pragma unroll
    for (int kc = 0; kc < 4; ++kc)
      bfrag[nt][kc] = load_b_frag(Whh16, H_, wave * 64 + nt * 16, kc * 32, lane);

  for (int i = tid; i < 16 * HSTR; i += 256) sh_h[i] = (_Float16)0.0f;

  // Elementwise ownership: row em, 8 cell columns starting at ek.
  const int em = tid >> 4;
  const int ek = (tid & 15) * 8;
  float creg[8];
#pragma unroll
  for (int j = 0; j < 8; ++j) creg[j] = 0.0f;

  const int nlane = lane & 15;
  const int mHalf = (lane >> 4) ? 8 : 0;

#if USE_ASYNC_XG
  // Per-thread async-copy slice of an xg step tile: row xrow, 4 x b128 chunks.
  const int xrow = tid >> 4;            // 0..15
  const int xcol = (tid & 15) * 32;     // base half-index; chunks at +0,8,16,24
  // Prologue: prefetch step 0 into buffer 0.
  {
    const _Float16* g0 = xg + ((size_t)0 * B_ + wgRow + xrow) * G4_ + xcol;
    _Float16* l0 = &sh_xg[0][xrow * XSTR + xcol];
#pragma unroll
    for (int q = 0; q < 4; ++q)
      async_copy_b128(g0 + q * 8, l0 + q * 8);
  }
#endif

  __syncthreads();

  for (int t = 0; t < T_; ++t) {
#if USE_ASYNC_XG
    // Prefetch xg for step t+1 (wrap at the end; harmless, keeps waits constant).
    {
      const int t1 = (t + 1) & (T_ - 1);
      const _Float16* gs = xg + ((size_t)t1 * B_ + wgRow + xrow) * G4_ + xcol;
      _Float16* ls = &sh_xg[(t + 1) & 1][xrow * XSTR + xcol];
#pragma unroll
      for (int q = 0; q < 4; ++q)
        async_copy_b128(gs + q * 8, ls + q * 8);
    }
#else
    // Fallback software pipeline: issue this step's xg loads before the GEMM.
    const _Float16* xgp = xg + ((size_t)t * B_ + (wgRow + em)) * G4_;
    v8h xi = *(const v8h*)(xgp + ek);
    v8h xf = *(const v8h*)(xgp + 128 + ek);
    v8h xc = *(const v8h*)(xgp + 256 + ek);
    v8h xo = *(const v8h*)(xgp + 384 + ek);
#endif

    // ---- GEMM: gates = h @ Whh^T ----
    v16h a[4];
#pragma unroll
    for (int kc = 0; kc < 4; ++kc)
      a[kc] = load_a_frag(sh_h, HSTR, kc * 32, lane);
#pragma unroll
    for (int nt = 0; nt < 4; ++nt) {
      v8f c = {};
#pragma unroll
      for (int kc = 0; kc < 4; ++kc)
        c = __builtin_amdgcn_wmma_f32_16x16x32_f16(false, a[kc], false,
                                                   bfrag[nt][kc], (short)0, c,
                                                   false, false);
      const int n = wave * 64 + nt * 16 + nlane;
#pragma unroll
      for (int r = 0; r < 8; ++r)
        sh_g[(mHalf + r) * GSTR + n] = (_Float16)c[r];
    }

#if USE_ASYNC_XG
    // Retire this step's xg prefetch (the 4 just-issued t+1 loads may remain).
    __builtin_amdgcn_s_wait_asynccnt(4);
#endif
    __syncthreads();

    // ---- LSTM cell elementwise (i,f,g,o PyTorch order) ----
#if USE_ASYNC_XG
    const _Float16* xr = &sh_xg[t & 1][em * XSTR];
    v8h xi = *(const v8h*)(xr + ek);
    v8h xf = *(const v8h*)(xr + 128 + ek);
    v8h xc = *(const v8h*)(xr + 256 + ek);
    v8h xo = *(const v8h*)(xr + 384 + ek);
#endif
    const _Float16* gr = sh_g + em * GSTR;
    v8h gvi = *(const v8h*)(gr + ek);
    v8h gvf = *(const v8h*)(gr + 128 + ek);
    v8h gvc = *(const v8h*)(gr + 256 + ek);
    v8h gvo = *(const v8h*)(gr + 384 + ek);
    v8h hout;
#pragma unroll
    for (int j = 0; j < 8; ++j) {
      float gi = (float)gvi[j] + (float)xi[j];
      float gf = (float)gvf[j] + (float)xf[j];
      float gc = (float)gvc[j] + (float)xc[j];
      float go = (float)gvo[j] + (float)xo[j];
      float cc = fast_sigmoid(gf) * creg[j] + fast_sigmoid(gi) * fast_tanh(gc);
      creg[j] = cc;
      hout[j] = (_Float16)(fast_sigmoid(go) * fast_tanh(cc));
    }
    __syncthreads();   // all gate reads & A-frag reads done

    *(v8h*)&sh_h[em * HSTR + ek] = hout;
    *(v8h*)&hbuf[((size_t)t * B_ + (wgRow + em)) * H_ + ek] = hout;
    __syncthreads();   // h visible for next step
  }
}

// ---------------------------------------------------------------------------
// Final FC: out[b,o] = h_T[b,:] . Wfc[o,:] + bfc[o]   (256x3)
// ---------------------------------------------------------------------------
__global__ void fc_kernel(const _Float16* __restrict__ hbuf,
                          const float* __restrict__ Wfc,
                          const float* __restrict__ bfc,
                          float* __restrict__ out) {
  int idx = blockIdx.x * blockDim.x + threadIdx.x;
  if (idx >= B_ * 3) return;
  int b = idx / 3, o = idx % 3;
  const _Float16* h = hbuf + ((size_t)(T_ - 1) * B_ + b) * H_;
  const float* w = Wfc + o * H_;
  float acc = bfc[o];
#pragma unroll 4
  for (int k = 0; k < H_; ++k) acc = fmaf((float)h[k], w[k], acc);
  out[b * 3 + o] = acc;
}

// ---------------------------------------------------------------------------
extern "C" void kernel_launch(void* const* d_in, const int* in_sizes, int n_in,
                              void* d_out, int out_size, void* d_ws, size_t ws_size,
                              hipStream_t stream) {
  const float* x    = (const float*)d_in[0];
  const float* Wih[3] = {(const float*)d_in[1], (const float*)d_in[5], (const float*)d_in[9]};
  const float* Whh[3] = {(const float*)d_in[2], (const float*)d_in[6], (const float*)d_in[10]};
  const float* bih[3] = {(const float*)d_in[3], (const float*)d_in[7], (const float*)d_in[11]};
  const float* bhh[3] = {(const float*)d_in[4], (const float*)d_in[8], (const float*)d_in[12]};
  const float* Wfc = (const float*)d_in[13];
  const float* bfc = (const float*)d_in[14];
  float* out = (float*)d_out;

  char* ws = (char*)d_ws;
  const size_t XG_BYTES = (size_t)T_ * B_ * G4_ * sizeof(_Float16);  // 128 MB
  const size_t HB_BYTES = (size_t)T_ * B_ * H_  * sizeof(_Float16);  // 32 MB
  _Float16* xg   = (_Float16*)(ws);
  _Float16* hbuf = (_Float16*)(ws + XG_BYTES);
  char* wp = ws + XG_BYTES + HB_BYTES;
  _Float16* whh16[3];
  for (int l = 0; l < 3; ++l) { whh16[l] = (_Float16*)wp; wp += (size_t)G4_ * H_ * 2; }
  _Float16* wih16_12[2];
  for (int l = 0; l < 2; ++l) { wih16_12[l] = (_Float16*)wp; wp += (size_t)G4_ * H_ * 2; }
  float* bias[3];
  for (int l = 0; l < 3; ++l) { bias[l] = (float*)wp; wp += (size_t)G4_ * 4; }

  // ---- weight prep ----
  const int WN = G4_ * H_;                       // 65536
  for (int l = 0; l < 3; ++l)
    cvt_f32_to_f16<<<(WN + 255) / 256, 256, 0, stream>>>(Whh[l], whh16[l], WN);
  cvt_f32_to_f16<<<(WN + 255) / 256, 256, 0, stream>>>(Wih[1], wih16_12[0], WN);
  cvt_f32_to_f16<<<(WN + 255) / 256, 256, 0, stream>>>(Wih[2], wih16_12[1], WN);
  for (int l = 0; l < 3; ++l)
    bias_sum_kernel<<<2, 256, 0, stream>>>(bih[l], bhh[l], bias[l], G4_);

  // ---- layer 0 ----
  {
    size_t n = (size_t)T_ * B_ * 128;
    xg_l0_kernel<<<(int)((n + 255) / 256), 256, 0, stream>>>(x, Wih[0], bias[0], xg);
    lstm_scan_kernel<<<B_ / 16, 256, 0, stream>>>(xg, whh16[0], hbuf);
  }
  // ---- layers 1,2 ----
  for (int l = 1; l < 3; ++l) {
    xg_gemm_kernel<<<(T_ * B_) / 64, 256, 0, stream>>>(hbuf, wih16_12[l - 1], bias[l], xg);
    lstm_scan_kernel<<<B_ / 16, 256, 0, stream>>>(xg, whh16[l], hbuf);
  }
  // ---- FC head ----
  fc_kernel<<<3, 256, 0, stream>>>(hbuf, Wfc, bfc, out);
}
